// RoIAlignExemplarEmbedder_24163486007435
// MI455X (gfx1250) — compile-verified, matrix operand order checked by reading, not run on hardware
//
#include <hip/hip_runtime.h>
#include <hip/hip_bf16.h>

typedef unsigned short u16;
typedef unsigned int   u32;
typedef __attribute__((ext_vector_type(16))) __bf16 v16bf;
typedef __attribute__((ext_vector_type(8)))  float  v8f;

// ---------- helpers ----------
__device__ __forceinline__ u16 f2bf(float f) {
    u32 x = __float_as_uint(f);
    u32 r = x + 0x7FFFu + ((x >> 16) & 1u);   // round-to-nearest-even
    return (u16)(r >> 16);
}
__device__ __forceinline__ float bf2f(u16 u) {
    return __uint_as_float(((u32)u) << 16);
}

// CDNA5 async global->LDS copy (ASYNCcnt-tracked, no VGPR round trip).
// lds_addr: wave-relative LDS byte offset (low 32 bits of the flat shared pointer).
__device__ __forceinline__ void async_load_b128(u32 lds_addr, const void* gaddr) {
    asm volatile("global_load_async_to_lds_b128 %0, %1, off"
                 :: "v"(lds_addr), "v"(gaddr) : "memory");
}
__device__ __forceinline__ void wait_async0() {
    asm volatile("s_wait_asynccnt 0" ::: "memory");
}
__device__ __forceinline__ u32 lds_off(const void* p) {
    return (u32)(uintptr_t)p;   // LDS aperture: addr[31:0] == LDS byte offset
}

// ---------- transpose + f32->bf16 convert ----------
// src: (batch, R, C) f32 row-major.  dst: (batch, C, R) bf16.  R,C multiples of 32.
__global__ __launch_bounds__(256) void transpose_cvt_kernel(
    const float* __restrict__ src, u16* __restrict__ dst, int R, int C)
{
    __shared__ float tile[32][33];
    size_t bofs = (size_t)blockIdx.z * (size_t)R * (size_t)C;
    int c0 = blockIdx.x * 32;
    int r0 = blockIdx.y * 32;
    int tx = threadIdx.x;         // 0..31
    int ty = threadIdx.y;         // 0..7
    #pragma unroll
    for (int j = 0; j < 32; j += 8)
        tile[ty + j][tx] = src[bofs + (size_t)(r0 + ty + j) * C + (c0 + tx)];
    __syncthreads();
    #pragma unroll
    for (int j = 0; j < 32; j += 8)
        dst[bofs + (size_t)(c0 + ty + j) * R + (r0 + tx)] = f2bf(tile[tx][ty + j]);
}

// ---------- RoIAlign 1x1 ----------
// zt: (bs, 64, 64, 1280) bf16. One block per ROI (grid=1024), 256 threads.
__global__ __launch_bounds__(256) void roi_align_kernel(
    const u16* __restrict__ zt, const float* __restrict__ bboxes,
    u16* __restrict__ xout)
{
    __shared__ float box[4];
    __shared__ int   ghs, gws;
    __shared__ float invcount;
    __shared__ int   yl[8], yh[8], xl[8], xh[8];
    __shared__ float fy[8], my[8], fx[8], mx[8];

    const int rid = blockIdx.x;
    const int b   = rid >> 7;          // rid / 128
    const int tid = threadIdx.x;

    if (tid < 4) box[tid] = bboxes[rid * 4 + tid] * 0.125f - 0.5f;
    __syncthreads();
    if (tid == 0) {
        float rw = box[2] - box[0], rh = box[3] - box[1];
        float gw = fminf(fmaxf(ceilf(rw), 1.0f), 8.0f);
        float gh = fminf(fmaxf(ceilf(rh), 1.0f), 8.0f);
        gws = (int)gw; ghs = (int)gh;
        invcount = 1.0f / (gw * gh);
    }
    __syncthreads();
    if (tid < 16) {
        int g = tid & 7;
        bool isy = tid < 8;
        float lo_c = isy ? box[1] : box[0];
        float hi_c = isy ? box[3] : box[2];
        int   gdim = isy ? ghs : gws;
        float c  = lo_c + (g + 0.5f) * (hi_c - lo_c) / (float)gdim;
        bool oob = (c < -1.0f) || (c > 64.0f);
        float cc = fmaxf(c, 0.0f);
        float l0 = floorf(cc);
        bool hic = (l0 >= 63.0f);
        int lo = (int)fminf(l0, 63.0f);
        int hi = (int)fminf(l0 + 1.0f, 63.0f);
        float fr = hic ? 0.0f : (cc - l0);
        float mk = (g < gdim && !oob) ? 1.0f : 0.0f;
        if (isy) { yl[g] = lo; yh[g] = hi; fy[g] = fr; my[g] = mk; }
        else     { xl[g] = lo; xh[g] = hi; fx[g] = fr; mx[g] = mk; }
    }
    __syncthreads();

    const size_t base = (size_t)b * 64 * 64 * 1280;
    for (int c = tid; c < 1280; c += 256) {
        float acc = 0.0f;
        for (int gy = 0; gy < ghs; ++gy) {
            float wy = my[gy];
            float hy = 1.0f - fy[gy], ly = fy[gy];
            size_t py0 = (size_t)(yl[gy] * 64), py1 = (size_t)(yh[gy] * 64);
            for (int gx = 0; gx < gws; ++gx) {
                float w = wy * mx[gx];
                float hx = 1.0f - fx[gx], lx = fx[gx];
                float v00 = bf2f(zt[base + (py0 + xl[gx]) * 1280 + c]);
                float v01 = bf2f(zt[base + (py0 + xh[gx]) * 1280 + c]);
                float v10 = bf2f(zt[base + (py1 + xl[gx]) * 1280 + c]);
                float v11 = bf2f(zt[base + (py1 + xh[gx]) * 1280 + c]);
                acc += w * (hy * (hx * v00 + lx * v01) + ly * (hx * v10 + lx * v11));
            }
        }
        xout[(size_t)rid * 1280 + c] = f2bf(acc * invcount);
    }
}

// ---------- tiny size-embedding MLP: sz(2) -> 64 -> 256, both ReLU ----------
__global__ __launch_bounds__(256) void size_embed_kernel(
    const float* __restrict__ bboxes,
    const float* __restrict__ S1, const float* __restrict__ sb1,
    const float* __restrict__ S2, const float* __restrict__ sb2,
    u16* __restrict__ e2out)
{
    __shared__ float szv[2];
    __shared__ float e1[64];
    const int rid = blockIdx.x;
    const int tid = threadIdx.x;
    if (tid == 0) {
        szv[0] = bboxes[rid * 4 + 3] - bboxes[rid * 4 + 1];  // h
        szv[1] = bboxes[rid * 4 + 2] - bboxes[rid * 4 + 0];  // w
    }
    __syncthreads();
    if (tid < 64) {
        float v = szv[0] * S1[tid] + szv[1] * S1[64 + tid] + sb1[tid];
        e1[tid] = fmaxf(v, 0.0f);
    }
    __syncthreads();
    float acc = sb2[tid];
    #pragma unroll 8
    for (int k = 0; k < 64; ++k) acc += e1[k] * S2[k * 256 + tid];
    e2out[(size_t)rid * 256 + tid] = f2bf(fmaxf(acc, 0.0f));
}

// ---------- WMMA bf16 GEMM:  C = A(MxK) * B(KxN) [+bias] [ReLU] ----------
// A: M x K bf16 row-major.  Bt: N x K bf16 row-major (B transposed).
// Block tile 128x128, 8 wave32 waves, each wave 64x32 = 4x2 v_wmma 16x16x32 tiles.
// Double-buffered LDS staging via CDNA5 async global->LDS loads (ASYNCcnt).
template<bool RELU, bool F32OUT>
__global__ __launch_bounds__(256) void gemm_bf16_wmma(
    const u16* __restrict__ A, const u16* __restrict__ Bt,
    const float* __restrict__ bias, void* __restrict__ Cout,
    int M, int N, int K, int ldc, int cofs)
{
    // 128 rows x 32 K, row stride 40 u16 (80B): 16B-aligned, conflict-free b128 frag reads
    __shared__ __align__(16) u16 As[2][128 * 40];
    __shared__ __align__(16) u16 Bs[2][128 * 40];

    const int tid  = threadIdx.x;
    const int lane = tid & 31;
    const int wave = tid >> 5;          // 8 waves (wave32)
    const int half = (lane >> 4) & 1;   // K-half select per ISA 16-bit A layout
    const int r    = lane & 15;
    const int wm   = wave >> 2;         // 0..1 -> 64-row strip
    const int wn   = wave & 3;          // 0..3 -> 32-col strip

    const int n0 = blockIdx.x * 128;
    const int m0 = blockIdx.y * 128;

    const v8f vzero = {0.f, 0.f, 0.f, 0.f, 0.f, 0.f, 0.f, 0.f};
    v8f acc[4][2];
    #pragma unroll
    for (int i = 0; i < 4; ++i)
        #pragma unroll
        for (int j = 0; j < 2; ++j) acc[i][j] = vzero;

    struct Frag { uint4 lo, hi; };

    // async-stage one 128x32 tile of A and Bt into LDS buffer `buf`
    auto stage = [&](int k0, int buf) {
        #pragma unroll
        for (int i = 0; i < 2; ++i) {
            int lin = tid + i * 256;          // 0..511
            int row = lin >> 2;               // 0..127
            int seg = (lin & 3) * 8;          // 0,8,16,24 (bf16 elems)
            async_load_b128(lds_off(&As[buf][row * 40 + seg]),
                            A + (size_t)(m0 + row) * K + k0 + seg);
            async_load_b128(lds_off(&Bs[buf][row * 40 + seg]),
                            Bt + (size_t)(n0 + row) * K + k0 + seg);
        }
    };

    stage(0, 0);
    wait_async0();
    __syncthreads();

    int buf = 0;
    for (int k0 = 0; k0 < K; k0 += 32) {
        const bool more = (k0 + 32) < K;
        if (more) stage(k0 + 32, buf ^ 1);   // DMA next tile while we compute

        if (more) {   // prefetch the tile after next toward L2/WGP$
            __builtin_prefetch(A  + (size_t)(m0 + (tid >> 1)) * K + k0 + 64, 0, 3);
            __builtin_prefetch(Bt + (size_t)(n0 + (tid >> 1)) * K + k0 + 64, 0, 3);
        }

        // Fragment loads (ISA 16-bit A layout: V0..3 = K[half*8 .. +7], V4..7 = K[16+half*8 .. +7])
        Frag fa[4], fb[2];
        #pragma unroll
        for (int mi = 0; mi < 4; ++mi) {
            int row = wm * 64 + mi * 16 + r;
            fa[mi].lo = *(const uint4*)(&As[buf][row * 40 + half * 8]);
            fa[mi].hi = *(const uint4*)(&As[buf][row * 40 + 16 + half * 8]);
        }
        #pragma unroll
        for (int ni = 0; ni < 2; ++ni) {
            int row = wn * 32 + ni * 16 + r;
            fb[ni].lo = *(const uint4*)(&Bs[buf][row * 40 + half * 8]);
            fb[ni].hi = *(const uint4*)(&Bs[buf][row * 40 + 16 + half * 8]);
        }
        #pragma unroll
        for (int mi = 0; mi < 4; ++mi) {
            v16bf av = __builtin_bit_cast(v16bf, fa[mi]);
            #pragma unroll
            for (int ni = 0; ni < 2; ++ni) {
                v16bf bv = __builtin_bit_cast(v16bf, fb[ni]);
                acc[mi][ni] = __builtin_amdgcn_wmma_f32_16x16x32_bf16(
                    false, av, false, bv, (short)0, acc[mi][ni], false, false);
            }
        }

        if (more) {
            wait_async0();      // next tile fully landed in LDS
            __syncthreads();    // all waves done reading `buf`, writes to buf^1 visible
        }
        buf ^= 1;
    }

    // epilogue: D layout — VGPR v holds (M = v + 8*half, N = r) within each 16x16 tile
    #pragma unroll
    for (int mi = 0; mi < 4; ++mi) {
        #pragma unroll
        for (int ni = 0; ni < 2; ++ni) {
            int col = n0 + wn * 32 + ni * 16 + r;
            float bv = bias[col];
            #pragma unroll
            for (int v = 0; v < 8; ++v) {
                int rowm = m0 + wm * 64 + mi * 16 + half * 8 + v;
                float val = acc[mi][ni][v] + bv;
                if (RELU) val = fmaxf(val, 0.0f);
                size_t oidx = (size_t)rowm * ldc + cofs + col;
                if (F32OUT) ((float*)Cout)[oidx] = val;
                else        ((u16*)Cout)[oidx]   = f2bf(val);
            }
        }
    }
}

// ---------- launch ----------
extern "C" void kernel_launch(void* const* d_in, const int* in_sizes, int n_in,
                              void* d_out, int out_size, void* d_ws, size_t ws_size,
                              hipStream_t stream) {
    const float* z      = (const float*)d_in[0];
    const float* bboxes = (const float*)d_in[1];
    const float* W1     = (const float*)d_in[2];
    const float* b1     = (const float*)d_in[3];
    const float* W2     = (const float*)d_in[4];
    const float* b2     = (const float*)d_in[5];
    const float* S1     = (const float*)d_in[6];
    const float* sb1    = (const float*)d_in[7];
    const float* S2     = (const float*)d_in[8];
    const float* sb2    = (const float*)d_in[9];
    const float* S3     = (const float*)d_in[10];
    const float* sb3    = (const float*)d_in[11];
    const float* O      = (const float*)d_in[12];
    const float* ob     = (const float*)d_in[13];
    float* out = (float*)d_out;

    (void)in_sizes; (void)n_in; (void)out_size; (void)ws_size;

    char* p = (char*)d_ws;
    auto alloc = [&](size_t bytes) -> char* {
        char* q = p; p += (bytes + 255) & ~(size_t)255; return q;
    };
    u16* W1t  = (u16*)alloc(5120ull * 1280 * 2);  // W1^T  (N=5120, K=1280)
    u16* W2t  = (u16*)alloc(1024ull * 5120 * 2);  // W2^T  (N=1024, K=5120)
    u16* Ot   = (u16*)alloc(1024ull * 2048 * 2);  // O^T   (N=1024, K=2048)
    u16* S3t  = (u16*)alloc(1024ull * 256  * 2);  // S3^T  (N=1024, K=256)
    u16* zt   = (u16*)alloc(8ull * 4096 * 1280 * 2);  // (b,H,W,ch) bf16
    u16* xA   = (u16*)alloc(1024ull * 1280 * 2);  // RoIAlign output (bf16)
    u16* h    = (u16*)alloc(1024ull * 5120 * 2);  // hidden after W1+ReLU
    u16* e2   = (u16*)alloc(1024ull * 256  * 2);  // size-embed hidden
    u16* xcat = (u16*)alloc(1024ull * 2048 * 2);  // concat [x2 | e3]

    dim3 tb(32, 8);
    // weight transposes + bf16 convert
    transpose_cvt_kernel<<<dim3(5120/32, 1280/32, 1), tb, 0, stream>>>(W1, W1t, 1280, 5120);
    transpose_cvt_kernel<<<dim3(1024/32, 5120/32, 1), tb, 0, stream>>>(W2, W2t, 5120, 1024);
    transpose_cvt_kernel<<<dim3(1024/32, 2048/32, 1), tb, 0, stream>>>(O,  Ot,  2048, 1024);
    transpose_cvt_kernel<<<dim3(1024/32,  256/32, 1), tb, 0, stream>>>(S3, S3t,  256, 1024);
    // z: (b, 1280, 4096) -> (b, 4096, 1280) bf16 for coalesced channel gathers
    transpose_cvt_kernel<<<dim3(4096/32, 1280/32, 8), tb, 0, stream>>>(z, zt, 1280, 4096);

    // RoIAlign and size embedding
    roi_align_kernel<<<1024, 256, 0, stream>>>(zt, bboxes, xA);
    size_embed_kernel<<<1024, 256, 0, stream>>>(bboxes, S1, sb1, S2, sb2, e2);

    // GEMM1: x(1024x1280) @ W1 -> h (1024x5120), +b1, ReLU
    gemm_bf16_wmma<true, false><<<dim3(5120/128, 1024/128), 256, 0, stream>>>(
        xA, W1t, b1, h, 1024, 5120, 1280, 5120, 0);
    // GEMM2: h @ W2 -> xcat[:, 0:1024], +b2
    gemm_bf16_wmma<false, false><<<dim3(1024/128, 1024/128), 256, 0, stream>>>(
        h, W2t, b2, xcat, 1024, 1024, 5120, 2048, 0);
    // GEMM3: e2(1024x256) @ S3 -> xcat[:, 1024:2048], +sb3
    gemm_bf16_wmma<false, false><<<dim3(1024/128, 1024/128), 256, 0, stream>>>(
        e2, S3t, sb3, xcat, 1024, 1024, 256, 2048, 1024);
    // GEMM4: xcat(1024x2048) @ O -> out f32 (1024x1024), +ob
    gemm_bf16_wmma<false, true><<<dim3(1024/128, 1024/128), 256, 0, stream>>>(
        xcat, Ot, ob, out, 1024, 1024, 2048, 1024, 0);
}
